// ProteinFeatures_48430051230499
// MI455X (gfx1250) — matrix operand order repaired and checked by reading, not compiled, not run on hardware
//
#include <hip/hip_runtime.h>
#include <math.h>

#define TOPK   30
#define NRBF   16
#define EF     128
#define NF     128
#define MAXREL 32
#define KIN    400          // EDGE_IN = 16*25
#define KPAD   416          // padded to 13 tiles of 32
#define KTILES 13

typedef __attribute__((ext_vector_type(16))) _Float16     v16h;
typedef __attribute__((ext_vector_type(8)))  float        v8f;
typedef __attribute__((ext_vector_type(8)))  unsigned int v8u;

union HU { v8u u; v16h h; };

// atom order: N=0, Ca=1, C=2, CB=3, O=4 ; 24 pairs after the D_neighbors RBF block
__device__ __constant__ int cPA[24] = {0,2,4,3, 1,1,1,1, 0,0,0, 3,3, 4, 0,2,4,3, 2,4,3, 2,4, 2};
__device__ __constant__ int cPB[24] = {0,2,4,3, 0,2,4,3, 2,4,3, 2,4, 2, 1,1,1,1, 0,0,0, 3,3, 4};

__device__ __forceinline__ void norm3(float x, float y, float z,
                                      float& ox, float& oy, float& oz) {
  float n = sqrtf(x*x + y*y + z*z);
  float inv = 1.0f / fmaxf(n, 1e-12f);
  ox = x*inv; oy = y*inv; oz = z*inv;
}
__device__ __forceinline__ void cross3(float ax,float ay,float az,
                                       float bx,float by,float bz,
                                       float& cx,float& cy,float& cz){
  cx = ay*bz - az*by; cy = az*bx - ax*bz; cz = ax*by - ay*bx;
}

// ---------------- backbone frames: bb (B,L,4,4) ----------------
__global__ void frames_bb_kernel(const float* __restrict__ X,
                                 float* __restrict__ bb, int BL) {
  int id = blockIdx.x*blockDim.x + threadIdx.x;
  if (id >= BL) return;
  const float* xa = X + (size_t)id*15;
  float Nx=xa[0],Ny=xa[1],Nz=xa[2];
  float Cax=xa[3],Cay=xa[4],Caz=xa[5];
  float Cx=xa[6],Cy=xa[7],Cz=xa[8];
  float v1x=Cx-Cax, v1y=Cy-Cay, v1z=Cz-Caz;
  float v2x=Nx-Cax, v2y=Ny-Cay, v2z=Nz-Caz;
  float e1x,e1y,e1z; norm3(v1x,v1y,v1z,e1x,e1y,e1z);
  float dp = e1x*v2x + e1y*v2y + e1z*v2z;
  float u2x=v2x-e1x*dp, u2y=v2y-e1y*dp, u2z=v2z-e1z*dp;
  float e2x,e2y,e2z; norm3(u2x,u2y,u2z,e2x,e2y,e2z);
  float e3x,e3y,e3z; cross3(e1x,e1y,e1z,e2x,e2y,e2z,e3x,e3y,e3z);
  float* o = bb + (size_t)id*16;
  o[0]=e1x; o[1]=e1y; o[2]=e1z;  o[3]=Cax;
  o[4]=e2x; o[5]=e2y; o[6]=e2z;  o[7]=Cay;
  o[8]=e3x; o[9]=e3y; o[10]=e3z; o[11]=Caz;
  o[12]=0.f; o[13]=0.f; o[14]=0.f; o[15]=0.f;
}

// ---------------- top-k neighbors: one wave32 per row (assumes L==1024) ------
__global__ void __launch_bounds__(128) topk_kernel(
    const float* __restrict__ X, const float* __restrict__ mask,
    float* __restrict__ eidx_out_f, float* __restrict__ dn_out,
    int* __restrict__ eidx_out_i, int B, int L) {
  const int lane = threadIdx.x & 31;
  const int wv   = threadIdx.x >> 5;
  int row = blockIdx.x*4 + wv;
  if (row >= B*L) return;
  int b = row / L, i = row % L;
  float mi  = mask[(size_t)b*L + i];
  const float* ci = X + (((size_t)b*L + i)*5 + 1)*3;
  float cix = ci[0], ciy = ci[1], ciz = ci[2];

  float dv[32]; float m2[32];
  float dmax = -1e30f;
  #pragma unroll
  for (int s = 0; s < 32; s++) {
    int j = lane + 32*s;
    const float* cj = X + (((size_t)b*L + j)*5 + 1)*3;
    float dx = cix-cj[0], dy = ciy-cj[1], dz = ciz-cj[2];
    float mm = mi * mask[(size_t)b*L + j];
    float d  = mm * sqrtf(dx*dx + dy*dy + dz*dz + 1e-6f);
    dv[s] = d; m2[s] = mm;
    dmax = fmaxf(dmax, d);
  }
  for (int off = 16; off; off >>= 1) dmax = fmaxf(dmax, __shfl_xor(dmax, off, 32));
  #pragma unroll
  for (int s = 0; s < 32; s++) dv[s] = dv[s] + (1.0f - m2[s])*dmax;

  for (int kk = 0; kk < TOPK; kk++) {
    float bv = 3.4e38f; int bs = 0;
    #pragma unroll
    for (int s = 0; s < 32; s++) if (dv[s] < bv) { bv = dv[s]; bs = s; }
    int bid = lane + 32*bs;                       // == neighbor index j
    for (int off = 16; off; off >>= 1) {
      float ov = __shfl_xor(bv, off, 32);
      int  oid = __shfl_xor(bid, off, 32);
      if (ov < bv || (ov == bv && oid < bid)) { bv = ov; bid = oid; }
    }
    if (lane == (bid & 31)) {
      int wslot = bid >> 5;
      #pragma unroll
      for (int s = 0; s < 32; s++) if (s == wslot) dv[s] = 3.4e38f;
    }
    if (lane == 0) {
      size_t e = (size_t)row*TOPK + kk;
      eidx_out_f[e] = (float)bid;
      eidx_out_i[e] = bid;
      dn_out[e]     = bv;            // D_neighbors == selected D_adjust
    }
  }
}

// ---- pack W_edge (400x128, f32) into WMMA B-fragment order, K padded to 416 --
__global__ void packW_kernel(const float* __restrict__ W_edge,
                             _Float16* __restrict__ packed) {
  int t = blockIdx.x*blockDim.x + threadIdx.x;
  if (t >= KTILES*8*32) return;
  int kt   = t / 256;
  int rem  = t % 256;
  int nt   = rem >> 5;
  int lane = rem & 31;
  int n     = nt*16 + (lane & 15);
  int kbase = kt*32 + ((lane >> 4) ? 16 : 0);
  for (int e = 0; e < 16; e++) {
    int K = kbase + e;
    float v = (K < KIN) ? W_edge[(size_t)K*EF + n] : 0.0f;
    packed[(size_t)t*16 + e] = (_Float16)v;
  }
}

// ---------------- node features: dihedrals -> 6x128 matmul + LN --------------
__global__ void __launch_bounds__(128) node_kernel(
    const float* __restrict__ X, const float* __restrict__ W_node,
    const float* __restrict__ g_node, const float* __restrict__ b_node,
    float* __restrict__ Vout, int B, int L) {
  int row = blockIdx.x;
  int b = row / L, i = row % L;
  int t = threadIdx.x;
  __shared__ float feat[6];
  __shared__ float wsum[4], wsq[4];
  if (t < 3) {
    int a = t;
    int q = 3*i + a - 1;
    float ang = 0.0f;
    if (q >= 0 && q <= 3*L - 4) {
      float p[4][3];
      #pragma unroll
      for (int s = 0; s < 4; s++) {
        int tt = q + s;
        const float* px = X + (((size_t)b*L + tt/3)*5 + (tt%3))*3;
        p[s][0]=px[0]; p[s][1]=px[1]; p[s][2]=px[2];
      }
      float u2x,u2y,u2z,u1x,u1y,u1z,u0x,u0y,u0z;
      norm3(p[1][0]-p[0][0], p[1][1]-p[0][1], p[1][2]-p[0][2], u2x,u2y,u2z);
      norm3(p[2][0]-p[1][0], p[2][1]-p[1][1], p[2][2]-p[1][2], u1x,u1y,u1z);
      norm3(p[3][0]-p[2][0], p[3][1]-p[2][1], p[3][2]-p[2][2], u0x,u0y,u0z);
      float a2x,a2y,a2z,a1x,a1y,a1z;
      cross3(u2x,u2y,u2z,u1x,u1y,u1z,a2x,a2y,a2z);
      cross3(u1x,u1y,u1z,u0x,u0y,u0z,a1x,a1y,a1z);
      float n2x,n2y,n2z,n1x,n1y,n1z;
      norm3(a2x,a2y,a2z,n2x,n2y,n2z);
      norm3(a1x,a1y,a1z,n1x,n1y,n1z);
      float cd = n2x*n1x + n2y*n1y + n2z*n1z;
      cd = fminf(fmaxf(cd, -1.0f + 1e-7f), 1.0f - 1e-7f);
      float sg = u2x*n1x + u2y*n1y + u2z*n1z;
      float sgn = (sg > 0.f) ? 1.f : ((sg < 0.f) ? -1.f : 0.f);
      ang = sgn * acosf(cd);
    }
    feat[a]   = cosf(ang);
    feat[3+a] = sinf(ang);
  }
  __syncthreads();
  float acc = 0.f;
  #pragma unroll
  for (int f = 0; f < 6; f++) acc += feat[f]*W_node[f*NF + t];
  float s1 = acc, s2 = acc*acc;
  for (int off = 16; off; off >>= 1) { s1 += __shfl_xor(s1, off, 32); s2 += __shfl_xor(s2, off, 32); }
  if ((t & 31) == 0) { wsum[t>>5] = s1; wsq[t>>5] = s2; }
  __syncthreads();
  float ts1 = wsum[0]+wsum[1]+wsum[2]+wsum[3];
  float ts2 = wsq[0]+wsq[1]+wsq[2]+wsq[3];
  float mean = ts1*(1.0f/128.0f);
  float var  = ts2*(1.0f/128.0f) - mean*mean;
  float rstd = rsqrtf(var + 1e-5f);
  Vout[(size_t)row*NF + t] = (acc - mean)*rstd*g_node[t] + b_node[t];
}

// -------- fused edge kernel: RBF features -> WMMA GEMM -> +E_pos -> LN -------
__global__ void __launch_bounds__(256) edge_kernel(
    const float* __restrict__ X, const float* __restrict__ W_pos,
    const float* __restrict__ b_pos, const float* __restrict__ g_edge,
    const float* __restrict__ b_edge, const _Float16* __restrict__ packedW,
    const float* __restrict__ Dn, const int* __restrict__ Eidx,
    float* __restrict__ Eout, int L, int nedges) {
  __shared__ _Float16 smA[16][KPAD];     // 16 edges x 416 features (f16)
  __shared__ float    smE[16][EF];       // post-GEMM tile (f32)
  __shared__ float    smAi[16][15];      // atoms of residue i
  __shared__ float    smAj[16][15];      // atoms of neighbor j
  __shared__ float    smDn[16];
  __shared__ int      smClip[16];
  __shared__ float    smMu[16], smRs[16];

  const int t = threadIdx.x;
  const size_t e0 = (size_t)blockIdx.x * 16;

  if (t < 16) {
    size_t edge = e0 + (size_t)t;
    if (edge >= (size_t)nedges) edge = (size_t)nedges - 1;
    int row = (int)(edge / TOPK);
    int i = row % L;
    int base_bi = row;                  // == b*L + i
    int j = Eidx[edge];
    smDn[t] = Dn[edge];
    int dc = i - j + MAXREL;
    dc = dc < 0 ? 0 : (dc > 2*MAXREL ? 2*MAXREL : dc);
    smClip[t] = dc;
    const float* xi = X + (size_t)base_bi*15;
    const float* xj = X + ((size_t)(row - i) + j)*15;   // (b*L + j)*15
    for (int c = 0; c < 15; c++) { smAi[t][c] = xi[c]; smAj[t][c] = xj[c]; }
  }
  __syncthreads();

  // feature build: 16 edges x 26 groups of 16 RBFs (group 25 = zero pad)
  for (int tk = t; tk < 16*26; tk += 256) {
    int m = tk / 26, g = tk % 26;
    if (g == 25) {
      for (int r = 0; r < 16; r++) smA[m][400 + r] = (_Float16)0.0f;
    } else {
      float d;
      if (g == 0) d = smDn[m];
      else {
        int p = g - 1; int pa = cPA[p], pb = cPB[p];
        float dx = smAi[m][pa*3+0] - smAj[m][pb*3+0];
        float dy = smAi[m][pa*3+1] - smAj[m][pb*3+1];
        float dz = smAi[m][pa*3+2] - smAj[m][pb*3+2];
        d = sqrtf(dx*dx + dy*dy + dz*dz + 1e-6f);
      }
      int base = g*16;
      #pragma unroll
      for (int r = 0; r < 16; r++) {
        float mu = 2.0f + (20.0f/15.0f)*(float)r;       // linspace(2,22,16)
        float z = (d - mu)*0.8f;                        // sigma = 1.25
        smA[m][base + r] = (_Float16)__expf(-z*z);
      }
    }
  }
  __syncthreads();

  // GEMM: 16(M edges) x 128(N) , K = 416 in 13 tiles of 32; wave wv owns N-tile
  const int lane = t & 31, wv = t >> 5;
  const int hi = lane >> 4, lo = lane & 15;
  v8f acc = {};
  const unsigned int* pw = (const unsigned int*)packedW;
  const unsigned short* arow = (const unsigned short*)&smA[lo][0];
  for (int kt = 0; kt < KTILES; kt++) {
    HU a, bfr;
    #pragma unroll
    for (int v = 0; v < 8; v++) {   // contiguous k order: VGPR v -> K pair 16*hi+2v, +1
      int kb = kt*32 + hi*16 + v*2;
      a.u[v] = *(const unsigned int*)(arow + kb);
    }
    bfr.u = *(const v8u*)(pw + ((size_t)((kt*8 + wv)*32 + lane))*8);
    acc = __builtin_amdgcn_wmma_f32_16x16x32_f16(
        false, a.h, false, bfr.h, (short)0, acc, false, false);
  }

  // epilogue: + E_pos (W_pos gather) + b_pos ; C/D layout: VGPR r -> M = r+8*hi
  #pragma unroll
  for (int r = 0; r < 8; r++) {
    int m = r + 8*hi;
    int n = wv*16 + lo;
    float val = acc[r] + W_pos[(size_t)smClip[m]*EF + n] + b_pos[n];
    smE[m][n] = val;
  }
  __syncthreads();

  // LayerNorm over 128 features per edge (16 lanes per edge)
  {
    int m = t >> 4, c = t & 15;
    float s1 = 0.f, s2 = 0.f;
    #pragma unroll
    for (int q = 0; q < 8; q++) { float x = smE[m][c + 16*q]; s1 += x; s2 += x*x; }
    for (int off = 8; off; off >>= 1) { s1 += __shfl_xor(s1, off, 32); s2 += __shfl_xor(s2, off, 32); }
    if (c == 0) {
      float mean = s1*(1.0f/128.0f);
      float var  = s2*(1.0f/128.0f) - mean*mean;
      smMu[m] = mean; smRs[m] = rsqrtf(var + 1e-5f);
    }
  }
  __syncthreads();
  {
    int m = t >> 4, c = t & 15;
    if (e0 + (size_t)m < (size_t)nedges) {
      float mean = smMu[m], rs = smRs[m];
      size_t obase = (e0 + (size_t)m)*(size_t)EF;
      #pragma unroll
      for (int q = 0; q < 8; q++) {
        int n = c + 16*q;
        Eout[obase + n] = (smE[m][n] - mean)*rs*g_edge[n] + b_edge[n];
      }
    }
  }
}

extern "C" void kernel_launch(void* const* d_in, const int* in_sizes, int n_in,
                              void* d_out, int out_size, void* d_ws, size_t ws_size,
                              hipStream_t stream) {
  const float* X      = (const float*)d_in[0];
  const float* mask   = (const float*)d_in[1];
  /* d_in[2] Lvec unused (full length) */
  const float* W_edge = (const float*)d_in[3];
  const float* W_node = (const float*)d_in[4];
  const float* W_pos  = (const float*)d_in[5];
  const float* b_pos  = (const float*)d_in[6];
  const float* g_edge = (const float*)d_in[7];
  const float* b_edge = (const float*)d_in[8];
  const float* g_node = (const float*)d_in[9];
  const float* b_node = (const float*)d_in[10];

  const int B  = in_sizes[2];
  const int L  = in_sizes[1] / B;
  const int BL = B * L;
  const size_t edges = (size_t)BL * TOPK;

  // outputs concatenated flat in return order: V, E, E_idx, bb
  float* Vout  = (float*)d_out;
  float* Eout  = Vout + (size_t)BL * NF;
  float* EidxF = Eout + edges * EF;
  float* bbOut = EidxF + edges;

  // workspace: D_neighbors (f32), E_idx (int), packed W_edge (f16)
  char* ws = (char*)d_ws;
  float*    Dn      = (float*)ws;
  int*      EidxI   = (int*)(ws + edges * sizeof(float));
  _Float16* packedW = (_Float16*)(ws + 2 * edges * sizeof(float)); // 32B-aligned

  frames_bb_kernel<<<(BL + 255)/256, 256, 0, stream>>>(X, bbOut, BL);
  topk_kernel<<<(BL + 3)/4, 128, 0, stream>>>(X, mask, EidxF, Dn, EidxI, B, L);
  packW_kernel<<<(KTILES*8*32 + 255)/256, 256, 0, stream>>>(W_edge, packedW);
  node_kernel<<<BL, 128, 0, stream>>>(X, W_node, g_node, b_node, Vout, B, L);
  edge_kernel<<<(int)((edges + 15)/16), 256, 0, stream>>>(
      X, W_pos, b_pos, g_edge, b_edge, packedW, Dn, EidxI, Eout, L, (int)edges);
}
